// QConv2d_60284160967068
// MI455X (gfx1250) — 8-bit matrix operand order repaired and checked by reading, compiled, not run on hardware
//
#include <hip/hip_runtime.h>
#include <hip/hip_bf16.h>


typedef int v8i __attribute__((ext_vector_type(8)));

#define NB    16
#define CIN   32
#define COUT  32
#define HH    256
#define WW    256
#define IN_ZP 3
#define OUT_ZP (-5)

__global__ __launch_bounds__(256) void qconv2d_wmma_kernel(
    const int* __restrict__ x,
    const int* __restrict__ qw,
    const int* __restrict__ qbias,
    const float* __restrict__ wscale,
    float* __restrict__ out)
{
    __shared__ signed char lds_in[3 * 132 * 36];
    __shared__ signed char lds_w[10 * 32 * 36];
    __shared__ int lds_bias[COUT];
    __shared__ int lds_scale[COUT];
    __shared__ int lds_shift[COUT];

    const int tid = threadIdx.x;
    const int wt  = blockIdx.x & 1;
    const int h   = (blockIdx.x >> 1) & 255;
    const int b   = blockIdx.x >> 9;
    const int w0  = wt * 128;

    for (int i = tid; i < COUT * CIN * 9; i += 256) {
        const int co  = i / (CIN * 9);
        const int rem = i % (CIN * 9);
        const int ci  = rem / 9;
        const int t   = rem % 9;
        lds_w[(t * 32 + co) * 36 + ci] = (signed char)qw[i];
    }
    for (int i = tid; i < 32 * 36; i += 256)
        lds_w[288 * 36 + i] = 0;

    if (tid < COUT) {
        const int co = tid;
        int wsum = 0;
        for (int i = 0; i < CIN * 9; ++i) wsum += qw[co * CIN * 9 + i];
        lds_bias[co] = qbias[co] - wsum * IN_ZP;
        const float fs = 0.02f * wscale[co] / 0.05f;
        const float fb = floorf(log2f(127.0f / fs));
        lds_shift[co] = (int)fb;
        lds_scale[co] = (int)lrintf(fs * exp2f(fb));
    }

    for (int i = tid; i < 3 * CIN * 130; i += 256) {
        const int wl = i % 130;
        const int ci = (i / 130) % CIN;
        const int r  = i / (130 * CIN);
        const int row = h - 1 + r;
        const int col = w0 - 1 + wl;
        int v = IN_ZP;
        if (row >= 0 && row < HH && col >= 0 && col < WW)
            v = x[((b * CIN + ci) * HH + row) * WW + col];
        lds_in[(r * 132 + wl) * 36 + ci] = (signed char)v;
    }
    __syncthreads();

    const int lane = tid & 31;
    const int wave = tid >> 5;
    const int n    = lane & 15;
    const int hl   = lane >> 4;

    v8i acc0 = {0, 0, 0, 0, 0, 0, 0, 0};
    v8i acc1 = {0, 0, 0, 0, 0, 0, 0, 0};

    #pragma unroll
    for (int c = 0; c < 5; ++c) {
        v8i bf;
        #pragma unroll
        for (int v = 0; v < 8; ++v) {
            const int t   = 2 * c + (v >> 2);
            const int cin = hl * 16 + (v & 3) * 4;
            int val = 0;
            if (t < 9) {
                const int kh = t / 3, kw = t % 3;
                const int wl = wave * 16 + n + kw;
                val = *(const int*)&lds_in[(kh * 132 + wl) * 36 + cin];
            }
            bf[v] = val;
        }

        #pragma unroll
        for (int half = 0; half < 2; ++half) {
            v8i af;
            #pragma unroll
            for (int v = 0; v < 8; ++v) {
                const int g    = (v & 3) + 4 * hl + ((v & 4) ? 8 : 0);
                const int t    = 2 * c + (g >> 3);
                const int cinb = (g & 7) * 4;
                const int co   = half * 16 + n;
                af[v] = *(const int*)&lds_w[(t * 32 + co) * 36 + cinb];
            }
            if (half == 0)
                acc0 = __builtin_amdgcn_wmma_i32_16x16x64_iu8(true, af, true, bf, acc0, false, false);
            else
                acc1 = __builtin_amdgcn_wmma_i32_16x16x64_iu8(true, af, true, bf, acc1, false, false);
        }
    }

    const int wout = w0 + wave * 16 + n;
    #pragma unroll
    for (int r = 0; r < 8; ++r) {
        const int m = r + 8 * hl;
        {
            const int co = m;
            int v = acc0[r] + lds_bias[co];
            v = (v * lds_scale[co]) >> lds_shift[co];
            v += OUT_ZP;
            v = v < -128 ? -128 : (v > 127 ? 127 : v);
            out[((b * COUT + co) * HH + h) * WW + wout] = (float)v;
        }
        {
            const int co = 16 + m;
            int v = acc1[r] + lds_bias[co];
            v = (v * lds_scale[co]) >> lds_shift[co];
            v += OUT_ZP;
            v = v < -128 ? -128 : (v > 127 ? 127 : v);
            out[((b * COUT + co) * HH + h) * WW + wout] = (float)v;
        }
    }
}

extern "C" void kernel_launch(void* const* d_in, const int* in_sizes, int n_in,
                              void* d_out, int out_size, void* d_ws, size_t ws_size,
                              hipStream_t stream) {
    const int*   x      = (const int*)d_in[0];
    const int*   qw     = (const int*)d_in[1];
    const int*   qbias  = (const int*)d_in[2];
    const float* wscale = (const float*)d_in[3];
    float*       out    = (float*)d_out;

    dim3 grid(NB * HH * 2);
    dim3 block(256);
    qconv2d_wmma_kernel<<<grid, block, 0, stream>>>(x, qw, qbias, wscale, out);
}
